// HGAPNet_76304388981322
// MI455X (gfx1250) — matrix operand order repaired and checked by reading, not compiled, not run on hardware
//
#include <hip/hip_runtime.h>

// ============================================================================
// HGAPNet forward for MI455X (gfx1250, wave32, WMMA 16x16x32 f16->f32).
// All dense linears run through one generic WMMA GEMM kernel (f16 in, f32
// accum, fused bias/ReLU/residual). Weights are staged in LDS per block
// (double-buffered, shared by 8 waves = 8 M-tiles); activations stream from
// L2 with prefetch. Activations staged f16, K padded to multiples of 32.
// Irregular ops (kNN, softmax-attention, gathers, pooling, interpolation)
// are wave32 VALU kernels. Whole working set is L2-resident (<200MB).
// ============================================================================

typedef __attribute__((ext_vector_type(16))) _Float16 v16h;
typedef __attribute__((ext_vector_type(8)))  _Float16 v8h;
typedef __attribute__((ext_vector_type(8)))  float    v8f;

static inline int eg(long long total) { return (int)((total + 255) / 256); }

// ---------------------------------------------------------------------------
// Generic WMMA GEMM: C[MxN] = act(A[MxKp] @ W[KpxN] + bias) (+ residual R)
// Block = 256 threads = 8 waves = 8 consecutive M-tiles sharing one N-group
// of NT 16-col subtiles. Weight K-chunks (32 x 16*NT f16) are staged in LDS,
// double-buffered: next chunk's global->LDS copy overlaps current WMMAs.
// Fragment layouts per CDNA5 ISA 7.12.2:
//   A (16x32 f16): lane = M%16; elems 0..7 = K base..base+7, 8..15 = base+16..23,
//                  base = (lane>=16 ? 8 : 0)
//   B (32x16 f16): lane%16 = N; elems = K, same order as A
//   C/D (16x16 f32): VGPR r, lane L: M = r + 8*(L>=16), N = L%16
// ---------------------------------------------------------------------------
template <int NT>
__global__ void __launch_bounds__(256)
k_wmma_gemm(const _Float16* __restrict__ A, const _Float16* __restrict__ W,
            const float* __restrict__ bias, const _Float16* __restrict__ R,
            _Float16* __restrict__ Ch, float* __restrict__ Cf,
            int M, int Kp, int N, int do_relu, int nb) {
  constexpr int CW  = 16 * NT;          // chunk width in halves
  constexpr int EPT = (32 * CW) / 256;  // chunk elements per thread (= 2*NT)
  __shared__ __align__(32) _Float16 sw[2][32 * CW];

  const int tid  = threadIdx.x;
  const int lane = tid & 31;
  const int wv   = tid >> 5;
  const int bm   = blockIdx.x / nb;     // M-block (8 tiles)
  const int gn   = blockIdx.x % nb;     // N-group
  const int tm   = bm * 8 + wv;
  const int tn0  = gn * NT;
  const bool active = (tm * 16 < M);
  const int mrow  = active ? (tm * 16 + (lane & 15)) : (lane & 15);
  const int abase = (lane >> 4) << 3;

  v8f acc[NT];
#pragma unroll
  for (int t = 0; t < NT; ++t) {
    float b = bias[(tn0 + t) * 16 + (lane & 15)];
#pragma unroll
    for (int r = 0; r < 8; ++r) acc[t][r] = b;
  }

  // stage weight chunk 0
  {
    const _Float16* gw = W + gn * CW;
#pragma unroll
    for (int e = 0; e < EPT; ++e) {
      int el = tid * EPT + e;
      int r = el / CW, c = el % CW;
      sw[0][el] = gw[(size_t)r * N + c];
    }
  }
  __syncthreads();

  int kc = 0;
  for (int k0 = 0; k0 < Kp; k0 += 32) {
    // overlap: stage next weight chunk into the other buffer
    if (k0 + 32 < Kp) {
      const _Float16* gw = W + (size_t)(k0 + 32) * N + gn * CW;
#pragma unroll
      for (int e = 0; e < EPT; ++e) {
        int el = tid * EPT + e;
        int r = el / CW, c = el % CW;
        sw[kc ^ 1][el] = gw[(size_t)r * N + c];
      }
    }

    const _Float16* ap = A + (size_t)mrow * Kp + k0 + abase;
    __builtin_prefetch((const void*)(ap + 32), 0, 1);  // next A K-chunk
    v8h alo = *(const v8h*)(ap);
    v8h ahi = *(const v8h*)(ap + 16);
    v16h afrag;
#pragma unroll
    for (int j = 0; j < 8; ++j) { afrag[j] = alo[j]; afrag[j + 8] = ahi[j]; }

    const _Float16* scol = &sw[kc][lane & 15];  // lane%16 = N; elems = K (same order as A)
#pragma unroll
    for (int t = 0; t < NT; ++t) {
      v16h bfrag;
#pragma unroll
      for (int j = 0; j < 8; ++j) {
        bfrag[j]     = scol[(abase + j) * CW + t * 16];
        bfrag[j + 8] = scol[(abase + 16 + j) * CW + t * 16];
      }
      acc[t] = __builtin_amdgcn_wmma_f32_16x16x32_f16(
          /*neg_a=*/false, afrag, /*neg_b=*/false, bfrag,
          /*c_mod=*/(short)0, acc[t], /*reuse_a=*/false, /*reuse_b=*/false);
    }
    __syncthreads();
    kc ^= 1;
  }

  if (!active) return;
#pragma unroll
  for (int t = 0; t < NT; ++t) {
#pragma unroll
    for (int r = 0; r < 8; ++r) {
      int mm = tm * 16 + r + ((lane >> 4) << 3);
      int nn = (tn0 + t) * 16 + (lane & 15);
      float v = acc[t][r];
      if (do_relu) v = fmaxf(v, 0.0f);
      if (R)  v += (float)R[(size_t)mm * N + nn];
      if (Ch) Ch[(size_t)mm * N + nn] = (_Float16)v;
      if (Cf) Cf[(size_t)mm * N + nn] = v;
    }
  }
}

// ---- weight fp32 -> fp16 with K padded to Kp (zero tail) -------------------
__global__ void k_cvt_w(const float* __restrict__ src, _Float16* __restrict__ dst,
                        int K, int N, int Kp) {
  long long t = (long long)blockIdx.x * blockDim.x + threadIdx.x;
  if (t >= (long long)Kp * N) return;
  int k = (int)(t / N), c = (int)(t % N);
  dst[t] = (k < K) ? (_Float16)src[(size_t)k * N + c] : (_Float16)0.0f;
}

// ---- mesh encoder input: (N,4,3) feat + (4,8) token -> (4N, 32) f16 --------
__global__ void k_mesh_in(const float* __restrict__ feat, const float* __restrict__ tok,
                          _Float16* __restrict__ A, int Npts) {
  long long t = (long long)blockIdx.x * blockDim.x + threadIdx.x;
  if (t >= (long long)Npts * 4 * 32) return;
  int r = (int)(t / 32), c = (int)(t % 32);
  int p = r >> 2, ti = r & 3;
  float v = 0.0f;
  if (c < 3)        v = feat[(size_t)p * 12 + ti * 3 + c];
  else if (c < 11)  v = tok[ti * 8 + (c - 3)];
  A[t] = (_Float16)v;
}

// ---- max over 4 tokens (color|normal) + pos concat -> (N, 96) f16 ----------
__global__ void k_cat_embed(const _Float16* __restrict__ hc, const _Float16* __restrict__ hn,
                            const float* __restrict__ pos, _Float16* __restrict__ A, int Npts) {
  long long t = (long long)blockIdx.x * blockDim.x + threadIdx.x;
  if (t >= (long long)Npts * 96) return;
  int p = (int)(t / 96), c = (int)(t % 96);
  float v = 0.0f;
  if (c < 32) {
    float mx = -3.0e38f;
    for (int j = 0; j < 4; ++j) mx = fmaxf(mx, (float)hc[((size_t)p * 4 + j) * 32 + c]);
    v = mx;
  } else if (c < 64) {
    float mx = -3.0e38f;
    for (int j = 0; j < 4; ++j) mx = fmaxf(mx, (float)hn[((size_t)p * 4 + j) * 32 + (c - 32)]);
    v = mx;
  } else if (c < 67) {
    v = pos[(size_t)p * 3 + (c - 64)];
  }
  A[t] = (_Float16)v;
}

// ---- strided row pooling (perm = arange(0, n, 4)) --------------------------
__global__ void k_pool_f32(const float* __restrict__ src, float* __restrict__ dst,
                           int m, int w) {
  long long t = (long long)blockIdx.x * blockDim.x + threadIdx.x;
  if (t >= (long long)m * w) return;
  int r = (int)(t / w), c = (int)(t % w);
  dst[t] = src[((size_t)r * 4) * w + c];
}
__global__ void k_pool_f16(const _Float16* __restrict__ src, _Float16* __restrict__ dst,
                           int m, int w) {
  long long t = (long long)blockIdx.x * blockDim.x + threadIdx.x;
  if (t >= (long long)m * w) return;
  int r = (int)(t / w), c = (int)(t % w);
  dst[t] = src[((size_t)r * 4) * w + c];
}

// ---- brute-force kNN: one thread per query, register-resident top-K --------
template <int K>
__global__ void k_knn(const float* __restrict__ qp, const float* __restrict__ sp,
                      int m, int n, int* __restrict__ idx, float* __restrict__ d2o) {
  int q = blockIdx.x * blockDim.x + threadIdx.x;
  if (q >= m) return;
  float qx = qp[3 * q + 0], qy = qp[3 * q + 1], qz = qp[3 * q + 2];
  float bd[K]; int bi[K];
#pragma unroll
  for (int j = 0; j < K; ++j) { bd[j] = 3.0e38f; bi[j] = 0; }
  for (int i = 0; i < n; ++i) {
    float dx = sp[3 * i + 0] - qx;
    float dy = sp[3 * i + 1] - qy;
    float dz = sp[3 * i + 2] - qz;
    float d = fmaf(dx, dx, fmaf(dy, dy, dz * dz));
    int wj = 0; float wv = bd[0];
#pragma unroll
    for (int j = 1; j < K; ++j) if (bd[j] > wv) { wv = bd[j]; wj = j; }
    if (d < wv) {
#pragma unroll
      for (int j = 0; j < K; ++j) if (j == wj) { bd[j] = d; bi[j] = i; }
    }
  }
#pragma unroll
  for (int j = 0; j < K; ++j) {
    idx[(size_t)q * K + j] = bi[j];
    if (d2o) d2o[(size_t)q * K + j] = bd[j];
  }
}

// ---- row gather ------------------------------------------------------------
__global__ void k_gather(const _Float16* __restrict__ src, const int* __restrict__ idx,
                         _Float16* __restrict__ dst, long long rows, int ch) {
  long long t = (long long)blockIdx.x * blockDim.x + threadIdx.x;
  if (t >= rows * ch) return;
  long long r = t / ch; int c = (int)(t % ch);
  dst[t] = src[(size_t)idx[r] * ch + c];
}

// ---- rel = pos[idx] - pos_pool[q], padded to width 32 ----------------------
__global__ void k_rel(const float* __restrict__ pos, const float* __restrict__ pp,
                      const int* __restrict__ idx, _Float16* __restrict__ rel,
                      long long rows) {
  long long t = (long long)blockIdx.x * blockDim.x + threadIdx.x;
  if (t >= rows * 32) return;
  long long r = t / 32; int c = (int)(t % 32);
  float v = 0.0f;
  if (c < 3) v = pos[(size_t)idx[r] * 3 + c] - pp[(size_t)(r >> 4) * 3 + c];
  rel[t] = (_Float16)v;
}

// ---- s = q[qidx] - k_gathered + pe ----------------------------------------
__global__ void k_s(const _Float16* __restrict__ qv, const _Float16* __restrict__ kg,
                    const _Float16* __restrict__ pe, _Float16* __restrict__ s,
                    long long total, int ch) {
  long long t = (long long)blockIdx.x * blockDim.x + threadIdx.x;
  if (t >= total) return;
  long long r = t / ch; int c = (int)(t % ch);
  long long q = r >> 4;
  s[t] = (_Float16)((float)qv[(size_t)q * ch + c] - (float)kg[t] + (float)pe[t]);
}

// ---- softmax over 16 neighbors + weighted sum of (v+pe) + residual ---------
__global__ void k_attn(const _Float16* __restrict__ s2, const _Float16* __restrict__ vg,
                       const _Float16* __restrict__ pe, const _Float16* __restrict__ xp,
                       _Float16* __restrict__ out, int m, int ch) {
  long long t = (long long)blockIdx.x * blockDim.x + threadIdx.x;
  if (t >= (long long)m * ch) return;
  int q = (int)(t / ch), c = (int)(t % ch);
  size_t base = (size_t)q * 16 * ch + c;
  float mx = -3.0e38f;
  for (int j = 0; j < 16; ++j) mx = fmaxf(mx, (float)s2[base + (size_t)j * ch]);
  float se = 0.0f, ac = 0.0f;
  for (int j = 0; j < 16; ++j) {
    size_t o = base + (size_t)j * ch;
    float e = __expf((float)s2[o] - mx);
    se += e;
    ac += e * ((float)vg[o] + (float)pe[o]);
  }
  out[t] = (_Float16)((float)xp[t] + ac / se);
}

// ---- inverse-distance (k=3) interpolate + skip concat ----------------------
__global__ void k_interp_cat(const _Float16* __restrict__ xskip, const _Float16* __restrict__ xc,
                             const int* __restrict__ idx3, const float* __restrict__ d23,
                             _Float16* __restrict__ A, int nf, int chf, int chc) {
  int chin = chf + chc;
  long long t = (long long)blockIdx.x * blockDim.x + threadIdx.x;
  if (t >= (long long)nf * chin) return;
  int p = (int)(t / chin), c = (int)(t % chin);
  float v;
  if (c < chf) {
    v = (float)xskip[(size_t)p * chf + c];
  } else {
    int cc = c - chf;
    float ws = 0.0f, acv = 0.0f;
    for (int j = 0; j < 3; ++j) {
      float w = 1.0f / (d23[(size_t)p * 3 + j] + 1e-8f);
      ws += w;
      acv += w * (float)xc[(size_t)idx3[(size_t)p * 3 + j] * chc + cc];
    }
    v = acv / ws;
  }
  A[t] = (_Float16)v;
}

// ===========================================================================
// Host side
// ===========================================================================
struct W16 { const _Float16* w; const float* b; int Kp; int N; };

static W16 make_w(void* const* d_in, int wi, int bi, int K, int N,
                  char* ws, size_t& off, hipStream_t s) {
  int Kp = (K + 31) & ~31;
  _Float16* dst = (_Float16*)(ws + off);
  off = (off + (size_t)Kp * N * sizeof(_Float16) + 255) & ~(size_t)255;
  k_cvt_w<<<eg((long long)Kp * N), 256, 0, s>>>((const float*)d_in[wi], dst, K, N, Kp);
  W16 r; r.w = dst; r.b = (const float*)d_in[bi]; r.Kp = Kp; r.N = N;
  return r;
}

static void gemm(const _Float16* A, const W16& w, const _Float16* R,
                 _Float16* Ch, float* Cf, int M, bool relu, hipStream_t s) {
  int tilesM = M / 16;
  int bm = (tilesM + 7) / 8;
  if (w.N % 64 == 0) {
    int nb = w.N / 64;
    k_wmma_gemm<4><<<bm * nb, 256, 0, s>>>(A, w.w, w.b, R, Ch, Cf, M, w.Kp, w.N, relu ? 1 : 0, nb);
  } else if (w.N % 32 == 0) {
    int nb = w.N / 32;
    k_wmma_gemm<2><<<bm * nb, 256, 0, s>>>(A, w.w, w.b, R, Ch, Cf, M, w.Kp, w.N, relu ? 1 : 0, nb);
  } else {
    int nb = w.N / 16;
    k_wmma_gemm<1><<<bm * nb, 256, 0, s>>>(A, w.w, w.b, R, Ch, Cf, M, w.Kp, w.N, relu ? 1 : 0, nb);
  }
}

extern "C" void kernel_launch(void* const* d_in, const int* in_sizes, int n_in,
                              void* d_out, int out_size, void* d_ws, size_t ws_size,
                              hipStream_t stream) {
  (void)in_sizes; (void)n_in; (void)out_size; (void)ws_size;
  char* ws = (char*)d_ws;
  size_t off = 0;  // bump allocator, reset every call -> deterministic

  auto aF16 = [&](size_t n) -> _Float16* {
    _Float16* p = (_Float16*)(ws + off);
    off = (off + n * sizeof(_Float16) + 255) & ~(size_t)255; return p;
  };
  auto aF32 = [&](size_t n) -> float* {
    float* p = (float*)(ws + off);
    off = (off + n * sizeof(float) + 255) & ~(size_t)255; return p;
  };
  auto aI32 = [&](size_t n) -> int* {
    int* p = (int*)(ws + off);
    off = (off + n * sizeof(int) + 255) & ~(size_t)255; return p;
  };

  const int N0 = 16384;
  const float* posf = (const float*)d_in[0];

  // ---- weights (pytree flatten: sorted dict keys, lin = {b,w}) -------------
  // 0 pos, 1 rgb, 2 normals, 3..5 color(lin.b, lin.w, tok),
  // 6..17 dec[i]{fuse.b,fuse.w,res.b,res.w}, 18..19 embed{b,w},
  // 20+18i enc[i]{g1,g2,pe1,pe2,res_blk,res_in,wk,wq,wv}x{b,w},
  // 74..79 head[0..2]{b,w}, 80..82 normal(lin.b, lin.w, tok)
  W16 w_color = make_w(d_in, 4, 3, 11, 32, ws, off, stream);
  W16 w_nrm   = make_w(d_in, 81, 80, 11, 32, ws, off, stream);
  W16 w_embed = make_w(d_in, 19, 18, 67, 64, ws, off, stream);
  W16 w_q[3], w_k[3], w_v[3], w_pe1[3], w_pe2[3], w_g1[3], w_g2[3], w_ri[3], w_rb[3];
  for (int i = 0; i < 3; ++i) {
    int ch = 64 << i, base = 20 + 18 * i;
    w_g1[i]  = make_w(d_in, base + 1,  base + 0,  ch,     ch,     ws, off, stream);
    w_g2[i]  = make_w(d_in, base + 3,  base + 2,  ch,     ch,     ws, off, stream);
    w_pe1[i] = make_w(d_in, base + 5,  base + 4,  3,      ch,     ws, off, stream);
    w_pe2[i] = make_w(d_in, base + 7,  base + 6,  ch,     ch,     ws, off, stream);
    w_rb[i]  = make_w(d_in, base + 9,  base + 8,  2 * ch, 2 * ch, ws, off, stream);
    w_ri[i]  = make_w(d_in, base + 11, base + 10, ch,     2 * ch, ws, off, stream);
    w_k[i]   = make_w(d_in, base + 13, base + 12, ch,     ch,     ws, off, stream);
    w_q[i]   = make_w(d_in, base + 15, base + 14, ch,     ch,     ws, off, stream);
    w_v[i]   = make_w(d_in, base + 17, base + 16, ch,     ch,     ws, off, stream);
  }
  W16 w_fuse[3], w_dres[3];
  {
    int up[4] = {512, 256, 128, 64};
    for (int i = 0; i < 3; ++i) {
      w_fuse[i] = make_w(d_in, 6 + 4 * i + 1, 6 + 4 * i + 0, up[i] + up[i + 1], up[i + 1], ws, off, stream);
      w_dres[i] = make_w(d_in, 6 + 4 * i + 3, 6 + 4 * i + 2, up[i + 1],         up[i + 1], ws, off, stream);
    }
  }
  W16 w_h0 = make_w(d_in, 75, 74, 64, 128, ws, off, stream);
  W16 w_h1 = make_w(d_in, 77, 76, 128, 128, ws, off, stream);
  W16 w_h2 = make_w(d_in, 79, 78, 128, 16, ws, off, stream);

  // ---- mesh encoders + embed ----------------------------------------------
  _Float16* Ac = aF16((size_t)N0 * 4 * 32);
  _Float16* An = aF16((size_t)N0 * 4 * 32);
  k_mesh_in<<<eg((long long)N0 * 4 * 32), 256, 0, stream>>>(
      (const float*)d_in[1], (const float*)d_in[5], Ac, N0);
  k_mesh_in<<<eg((long long)N0 * 4 * 32), 256, 0, stream>>>(
      (const float*)d_in[2], (const float*)d_in[82], An, N0);
  _Float16* Hc = aF16((size_t)N0 * 4 * 32);
  _Float16* Hn = aF16((size_t)N0 * 4 * 32);
  gemm(Ac, w_color, nullptr, Hc, nullptr, N0 * 4, true, stream);
  gemm(An, w_nrm,   nullptr, Hn, nullptr, N0 * 4, true, stream);
  _Float16* Xcat = aF16((size_t)N0 * 96);
  k_cat_embed<<<eg((long long)N0 * 96), 256, 0, stream>>>(Hc, Hn, posf, Xcat, N0);
  _Float16* X0 = aF16((size_t)N0 * 64);
  gemm(Xcat, w_embed, nullptr, X0, nullptr, N0, true, stream);

  // ---- encoder -------------------------------------------------------------
  const float* pos_lvl[4]; _Float16* x_lvl[4]; int n_lvl[4], ch_lvl[4];
  pos_lvl[0] = posf; x_lvl[0] = X0; n_lvl[0] = N0; ch_lvl[0] = 64;
  {
    const float* cp = posf; _Float16* cx = X0; int n = N0, ch = 64;
    for (int i = 0; i < 3; ++i) {
      int m = n / 4;
      float* pp = aF32((size_t)m * 3);
      k_pool_f32<<<eg((long long)m * 3), 256, 0, stream>>>(cp, pp, m, 3);
      _Float16* xp = aF16((size_t)m * ch);
      k_pool_f16<<<eg((long long)m * ch), 256, 0, stream>>>(cx, xp, m, ch);
      int* idx = aI32((size_t)m * 16);
      k_knn<16><<<eg(m), 256, 0, stream>>>(pp, cp, m, n, idx, nullptr);

      _Float16* qv = aF16((size_t)m * ch);
      gemm(xp, w_q[i], nullptr, qv, nullptr, m, false, stream);
      _Float16* kf = aF16((size_t)n * ch);
      gemm(cx, w_k[i], nullptr, kf, nullptr, n, false, stream);
      _Float16* vf = aF16((size_t)n * ch);
      gemm(cx, w_v[i], nullptr, vf, nullptr, n, false, stream);

      long long rows = (long long)m * 16;
      _Float16* kg = aF16((size_t)rows * ch);
      _Float16* vg = aF16((size_t)rows * ch);
      k_gather<<<eg(rows * ch), 256, 0, stream>>>(kf, idx, kg, rows, ch);
      k_gather<<<eg(rows * ch), 256, 0, stream>>>(vf, idx, vg, rows, ch);
      _Float16* rel = aF16((size_t)rows * 32);
      k_rel<<<eg(rows * 32), 256, 0, stream>>>(cp, pp, idx, rel, rows);

      _Float16* peh = aF16((size_t)rows * ch);
      gemm(rel, w_pe1[i], nullptr, peh, nullptr, (int)rows, true, stream);
      _Float16* pe = aF16((size_t)rows * ch);
      gemm(peh, w_pe2[i], nullptr, pe, nullptr, (int)rows, false, stream);

      _Float16* sM = aF16((size_t)rows * ch);
      k_s<<<eg(rows * ch), 256, 0, stream>>>(qv, kg, pe, sM, rows * ch, ch);
      _Float16* sh = aF16((size_t)rows * ch);
      gemm(sM, w_g1[i], nullptr, sh, nullptr, (int)rows, true, stream);
      _Float16* s2 = aF16((size_t)rows * ch);
      gemm(sh, w_g2[i], nullptr, s2, nullptr, (int)rows, false, stream);

      _Float16* hA = aF16((size_t)m * ch);
      k_attn<<<eg((long long)m * ch), 256, 0, stream>>>(s2, vg, pe, xp, hA, m, ch);

      _Float16* h2 = aF16((size_t)m * 2 * ch);
      gemm(hA, w_ri[i], nullptr, h2, nullptr, m, true, stream);
      _Float16* h3 = aF16((size_t)m * 2 * ch);
      gemm(h2, w_rb[i], h2, h3, nullptr, m, true, stream);  // x + relu(lin(x))

      cp = pp; cx = h3; n = m; ch *= 2;
      pos_lvl[i + 1] = pp; x_lvl[i + 1] = h3; n_lvl[i + 1] = n; ch_lvl[i + 1] = ch;
    }
  }

  // ---- decoder -------------------------------------------------------------
  _Float16* xi = x_lvl[3];
  for (int i = 0; i < 3; ++i) {
    int lc = 3 - i, lf = 2 - i;
    int nf = n_lvl[lf], chc = ch_lvl[lc], chf = ch_lvl[lf];
    int* idx3 = aI32((size_t)nf * 3);
    float* d23 = aF32((size_t)nf * 3);
    k_knn<3><<<eg(nf), 256, 0, stream>>>(pos_lvl[lf], pos_lvl[lc], nf, n_lvl[lc], idx3, d23);
    _Float16* Af = aF16((size_t)nf * (chf + chc));
    k_interp_cat<<<eg((long long)nf * (chf + chc)), 256, 0, stream>>>(
        x_lvl[lf], xi, idx3, d23, Af, nf, chf, chc);
    _Float16* t1 = aF16((size_t)nf * chf);
    gemm(Af, w_fuse[i], nullptr, t1, nullptr, nf, true, stream);
    _Float16* t2 = aF16((size_t)nf * chf);
    gemm(t1, w_dres[i], t1, t2, nullptr, nf, true, stream);
    xi = t2;
  }

  // ---- head ---------------------------------------------------------------
  _Float16* hh1 = aF16((size_t)N0 * 128);
  gemm(xi, w_h0, nullptr, hh1, nullptr, N0, true, stream);
  _Float16* hh2 = aF16((size_t)N0 * 128);
  gemm(hh1, w_h1, nullptr, hh2, nullptr, N0, true, stream);
  gemm(hh2, w_h2, nullptr, nullptr, (float*)d_out, N0, false, stream);
}
